// EncoderRNN_31198642438691
// MI455X (gfx1250) — compile-verified, matrix operand order checked by reading, not run on hardware
//
#include <hip/hip_runtime.h>
#include <hip/hip_fp16.h>
#include <math.h>

typedef __attribute__((ext_vector_type(16))) _Float16 v16h;
typedef __attribute__((ext_vector_type(8)))  _Float16 v8h;
typedef __attribute__((ext_vector_type(8)))  float    v8f;

#define B_    64
#define S_    512
#define E_    512
#define H_    512
#define NSEG_ 64
#define K_    512
#define G3H   1536   // 3*H
#define NCOLS 3072   // 2 dirs * 3H
#define NROWS 32768  // B*S
#define NBLK_SCAN 256

// ---------------------------------------------------------------------------
// WMMA fragment loader (works for global or LDS base after inlining).
// 16-bit A (16x32, MxK) / B (32x16, KxN) per-lane layout:
//   lane l<16 : row = l,    halves 0..7 = K[kb+0..7],  8..15 = K[kb+16..23]
//   lane l>=16: row = l-16, halves 0..7 = K[kb+8..15], 8..15 = K[kb+24..31]
// Caller passes row_base already offset by (lane&15)*ld.
// ---------------------------------------------------------------------------
__device__ __forceinline__ v16h frag16(const _Float16* __restrict__ row_base,
                                       int kb, int hi) {
  const _Float16* p = row_base + kb + hi * 8;
  v8h lo = *(const v8h*)(p);
  v8h hv = *(const v8h*)(p + 16);
  v16h r;
#pragma unroll
  for (int i = 0; i < 8; ++i) { r[i] = lo[i]; r[i + 8] = hv[i]; }
  return r;
}

// ---------------------------------------------------------------------------
// Init: zero parity-0 hidden state (f32 + f16) and the grid-barrier counter.
// ---------------------------------------------------------------------------
__global__ void k_init(float* __restrict__ hf32, _Float16* __restrict__ hf16,
                       unsigned* __restrict__ ctr) {
  int i = blockIdx.x * blockDim.x + threadIdx.x;
  const int n = B_ * H_;
  if (i < n) {
    hf32[(size_t)(0 * 2 + 0) * n + i] = 0.f;
    hf32[(size_t)(1 * 2 + 0) * n + i] = 0.f;
    hf16[(size_t)(0 * 2 + 0) * n + i] = (_Float16)0.f;
    hf16[(size_t)(1 * 2 + 0) * n + i] = (_Float16)0.f;
  }
  if (i == 0) *ctr = 0u;
}

// ---------------------------------------------------------------------------
// Embedding gather + f32->f16 convert.
// ---------------------------------------------------------------------------
__global__ void k_embed(const int* __restrict__ ids,
                        const float* __restrict__ emb,
                        _Float16* __restrict__ x16) {
  int row = blockIdx.x;
  int id  = ids[row];
  const float* src = emb + (size_t)id * E_;
  _Float16*    dst = x16 + (size_t)row * E_;
  for (int e = threadIdx.x; e < E_; e += blockDim.x)
    dst[e] = (_Float16)src[e];
}

__global__ void k_f32_to_f16(const float* __restrict__ src,
                             _Float16* __restrict__ dst, int n) {
  int i = blockIdx.x * blockDim.x + threadIdx.x;
  if (i < n) dst[i] = (_Float16)src[i];
}

// ---------------------------------------------------------------------------
// Input projection GEMM: xp[row][col] = x16[row][:] . W[col][:] + b_ih[col]
// M=32768, N=3072, K=512.  Block = 4 waves; wave = 16 rows x 64 cols.
// ---------------------------------------------------------------------------
__global__ __launch_bounds__(128) void k_xp_gemm(
    const _Float16* __restrict__ x16, const _Float16* __restrict__ w,
    const float* __restrict__ b_ih_f, const float* __restrict__ b_ih_b,
    float* __restrict__ xp) {
  const int lane = threadIdx.x & 31;
  const int wave = threadIdx.x >> 5;
  const int l15  = lane & 15;
  const int hi   = lane >> 4;
  const int row0 = blockIdx.x * 16;
  const int cb0  = blockIdx.y * 256 + wave * 64;

  const _Float16* arow = x16 + (size_t)(row0 + l15) * K_;
  const _Float16* brow[4];
#pragma unroll
  for (int t = 0; t < 4; ++t)
    brow[t] = w + (size_t)(cb0 + t * 16 + l15) * K_;

  v8f acc0 = {}, acc1 = {}, acc2 = {}, acc3 = {};
#pragma unroll
  for (int kb = 0; kb < K_; kb += 32) {
    if (kb + 64 < K_) __builtin_prefetch(arow + kb + 64, 0, 0);
    v16h a = frag16(arow, kb, hi);
    v16h b0 = frag16(brow[0], kb, hi);
    acc0 = __builtin_amdgcn_wmma_f32_16x16x32_f16(false, a, false, b0, (short)0, acc0, false, false);
    v16h b1 = frag16(brow[1], kb, hi);
    acc1 = __builtin_amdgcn_wmma_f32_16x16x32_f16(false, a, false, b1, (short)0, acc1, false, false);
    v16h b2 = frag16(brow[2], kb, hi);
    acc2 = __builtin_amdgcn_wmma_f32_16x16x32_f16(false, a, false, b2, (short)0, acc2, false, false);
    v16h b3 = frag16(brow[3], kb, hi);
    acc3 = __builtin_amdgcn_wmma_f32_16x16x32_f16(false, a, false, b3, (short)0, acc3, false, false);
  }
  v8f accs[4] = {acc0, acc1, acc2, acc3};
#pragma unroll
  for (int t = 0; t < 4; ++t) {
    int col = cb0 + t * 16 + l15;
    float bias = (col < G3H) ? b_ih_f[col] : b_ih_b[col - G3H];
#pragma unroll
    for (int r = 0; r < 8; ++r) {
      int m = r + 8 * hi;
      xp[(size_t)(row0 + m) * NCOLS + col] = accs[t][r] + bias;
    }
  }
}

// ---------------------------------------------------------------------------
// Persistent bidirectional GRU scan. 256 single-wave blocks =
// 2 dirs x 4 b-tiles(16) x 32 n-tiles(16 cols).
// Loop-invariant W_hh slice (3 gates x 16 cols x 512 k = 48KB) is staged into
// LDS once via async global->LDS copies (ASYNCcnt), then every step feeds
// WMMA B-fragments from ds_load.  Ping-pong h buffers; atomic grid barrier.
// ---------------------------------------------------------------------------
__global__ __launch_bounds__(32) void k_gru_scan(
    const float* __restrict__ xp, const _Float16* __restrict__ w_hh16,
    const float* __restrict__ b_hh_f, const float* __restrict__ b_hh_b,
    const int* __restrict__ lens, float* __restrict__ hf32,
    _Float16* __restrict__ hf16, float* __restrict__ outb,
    unsigned* __restrict__ ctr) {
  __shared__ _Float16 lw[3 * 16 * K_];   // [gate*16 + r][k], 48 KB

  const int lane = threadIdx.x;          // 0..31 (one wave)
  const int l15  = lane & 15;
  const int hi   = lane >> 4;
  const int bid  = blockIdx.x;
  const int d    = bid >> 7;                 // direction
  const int rid  = bid & 127;
  const int b0   = (rid >> 5) * 16;          // batch tile base
  const int jw   = (rid & 31) * 16;          // hidden col tile base
  const int jcol = jw + l15;

  // ---- stage this block's W_hh rows into LDS with async b128 copies ----
  // 3*16 rows x 512 halves = 3072 16-byte chunks; 96 per lane.
  {
#pragma unroll 4
    for (int i = 0; i < 96; ++i) {
      const int c      = lane + 32 * i;        // chunk id
      const int row_id = c >> 6;               // 0..47  (gate*16 + r)
      const int koff   = (c & 63) * 8;         // halves
      const int g      = row_id >> 4;
      const int r      = row_id & 15;
      const _Float16* gp = w_hh16 +
          (size_t)(d * G3H + g * H_ + jw + r) * K_ + koff;
      unsigned ldsoff = (unsigned)(uintptr_t)(&lw[row_id * K_ + koff]);
      unsigned long long ga = (unsigned long long)(uintptr_t)gp;
      asm volatile("global_load_async_to_lds_b128 %0, %1, off"
                   :: "v"(ldsoff), "v"(ga) : "memory");
    }
    asm volatile("s_wait_asynccnt 0x0" ::: "memory");
    __syncthreads();
  }

  const float* bhh = d ? b_hh_b : b_hh_f;
  const float bh_r = bhh[0 * H_ + jcol];
  const float bh_z = bhh[1 * H_ + jcol];
  const float bh_n = bhh[2 * H_ + jcol];

  const _Float16* wr = lw + (0 * 16 + l15) * K_;
  const _Float16* wz = lw + (1 * 16 + l15) * K_;
  const _Float16* wn = lw + (2 * 16 + l15) * K_;

  int Lr[8];
#pragma unroll
  for (int r = 0; r < 8; ++r) Lr[r] = lens[b0 + r + 8 * hi];

  const size_t slab = (size_t)B_ * H_;

  for (int t = 0; t < S_; ++t) {
    const int p = t & 1;
    const _Float16* hrd  = hf16 + (size_t)(d * 2 + p) * slab;
    const float*    hr32 = hf32 + (size_t)(d * 2 + p) * slab;
    float*          hw32 = hf32 + (size_t)(d * 2 + (1 - p)) * slab;
    _Float16*       hw16 = hf16 + (size_t)(d * 2 + (1 - p)) * slab;

    const _Float16* arow = hrd + (size_t)(b0 + l15) * H_;
    v8f cr = {}, cz = {}, cn = {};
#pragma unroll 4
    for (int kb = 0; kb < K_; kb += 32) {
      v16h a  = frag16(arow, kb, hi);       // global (h state)
      v16h fr = frag16(wr, kb, hi);         // LDS
      cr = __builtin_amdgcn_wmma_f32_16x16x32_f16(false, a, false, fr, (short)0, cr, false, false);
      v16h fz = frag16(wz, kb, hi);
      cz = __builtin_amdgcn_wmma_f32_16x16x32_f16(false, a, false, fz, (short)0, cz, false, false);
      v16h fn = frag16(wn, kb, hi);
      cn = __builtin_amdgcn_wmma_f32_16x16x32_f16(false, a, false, fn, (short)0, cn, false, false);
    }

#pragma unroll
    for (int r = 0; r < 8; ++r) {
      const int  bb = b0 + r + 8 * hi;
      const int  L  = Lr[r];
      const bool m  = (t < L);
      const int  trow = (d && m) ? (L - 1 - t) : t;  // backward: reversed gx row
      const float* gx = xp + ((size_t)bb * S_ + trow) * NCOLS + d * G3H;
      const float xr = gx[0 * H_ + jcol];
      const float xz = gx[1 * H_ + jcol];
      const float xn = gx[2 * H_ + jcol];
      const float hp = hr32[(size_t)bb * H_ + jcol];
      const float rg = 1.f / (1.f + __expf(-(xr + cr[r] + bh_r)));
      const float zg = 1.f / (1.f + __expf(-(xz + cz[r] + bh_z)));
      const float ng = tanhf(xn + rg * (cn[r] + bh_n));
      const float hn = (1.f - zg) * ng + zg * hp;
      const float ho = m ? hn : hp;
      hw32[(size_t)bb * H_ + jcol] = ho;
      hw16[(size_t)bb * H_ + jcol] = (_Float16)ho;
      outb[(((size_t)d * B_ + bb) * S_ + t) * H_ + jcol] = m ? hn : 0.f;
    }

    // ---- grid-wide barrier: all 256 blocks finish step t before t+1 ----
    __syncthreads();
    if (threadIdx.x == 0) {
      __threadfence();
      __hip_atomic_fetch_add(ctr, 1u, __ATOMIC_RELEASE, __HIP_MEMORY_SCOPE_AGENT);
      const unsigned target = (unsigned)(t + 1) * (unsigned)NBLK_SCAN;
      while (__hip_atomic_load(ctr, __ATOMIC_ACQUIRE, __HIP_MEMORY_SCOPE_AGENT) < target)
        __builtin_amdgcn_s_sleep(1);
      __threadfence();
    }
    __syncthreads();
  }
}

// ---------------------------------------------------------------------------
// Ragged segment max with backward un-reversal.
// ---------------------------------------------------------------------------
__global__ void k_segmax(const float* __restrict__ outb,
                         const int* __restrict__ layout,
                         const int* __restrict__ lens,
                         float* __restrict__ pooled) {
  const int b   = blockIdx.x >> 6;
  const int seg = blockIdx.x & (NSEG_ - 1);
  const int t0  = layout[b * (NSEG_ + 1) + seg];
  const int t1  = layout[b * (NSEG_ + 1) + seg + 1];
  const int L   = lens[b];
  for (int idx = threadIdx.x; idx < 2 * H_; idx += blockDim.x) {
    const int dir = idx >> 9;
    const int h   = idx & (H_ - 1);
    const float* base = outb + ((size_t)dir * B_ + b) * (size_t)S_ * H_;
    float acc = -3.402823466e+38f;
    for (int t = t0; t < t1; ++t) {
      const int tr = dir ? (L - 1 - t) : t;
      acc = fmaxf(acc, base[(size_t)tr * H_ + h]);
    }
    pooled[((size_t)b * NSEG_ + seg) * (2 * H_) + idx] = acc;
  }
}

// ---------------------------------------------------------------------------
// Final hidden state: after 512 steps the live parity is 0. [2,B,H] tail.
// ---------------------------------------------------------------------------
__global__ void k_copy_h(const float* __restrict__ hf32, float* __restrict__ dst) {
  int i = blockIdx.x * blockDim.x + threadIdx.x;
  const int n = B_ * H_;
  if (i < 2 * n) {
    int d = i / n, rem = i - d * n;
    dst[i] = hf32[(size_t)(d * 2 + 0) * n + rem];
  }
}

// ---------------------------------------------------------------------------
extern "C" void kernel_launch(void* const* d_in, const int* in_sizes, int n_in,
                              void* d_out, int out_size, void* d_ws, size_t ws_size,
                              hipStream_t stream) {
  (void)in_sizes; (void)n_in; (void)out_size; (void)ws_size;
  const int*   input_seqs = (const int*)d_in[0];
  const int*   input_lens = (const int*)d_in[1];
  const int*   layout     = (const int*)d_in[3];
  const float* emb_table  = (const float*)d_in[4];
  const float* w_ih_f     = (const float*)d_in[5];
  const float* w_hh_f     = (const float*)d_in[6];
  const float* b_ih_f     = (const float*)d_in[7];
  const float* b_hh_f     = (const float*)d_in[8];
  const float* w_ih_b     = (const float*)d_in[9];
  const float* w_hh_b     = (const float*)d_in[10];
  const float* b_ih_b     = (const float*)d_in[11];
  const float* b_hh_b     = (const float*)d_in[12];

  size_t off = 0;
  auto take = [&](size_t bytes) -> char* {
    char* p = (char*)d_ws + off;
    off += (bytes + 255) & ~(size_t)255;
    return p;
  };
  _Float16* x16    = (_Float16*)take((size_t)NROWS * K_ * 2);          //  32 MB
  _Float16* w_in   = (_Float16*)take((size_t)NCOLS * K_ * 2);          //   3 MB
  _Float16* w_hh16 = (_Float16*)take((size_t)2 * G3H * K_ * 2);        //   3 MB
  float*    xp     = (float*)take((size_t)NROWS * NCOLS * 4);          // 384 MB
  float*    outb   = (float*)take((size_t)2 * B_ * S_ * H_ * 4);       // 128 MB
  float*    hf32   = (float*)take((size_t)2 * 2 * B_ * H_ * 4);
  _Float16* hf16   = (_Float16*)take((size_t)2 * 2 * B_ * H_ * 2);
  unsigned* ctr    = (unsigned*)take(256);

  const int WCNT = G3H * K_;

  k_init<<<(B_ * H_ + 255) / 256, 256, 0, stream>>>(hf32, hf16, ctr);
  k_embed<<<NROWS, 256, 0, stream>>>(input_seqs, emb_table, x16);
  k_f32_to_f16<<<(WCNT + 255) / 256, 256, 0, stream>>>(w_ih_f, w_in, WCNT);
  k_f32_to_f16<<<(WCNT + 255) / 256, 256, 0, stream>>>(w_ih_b, w_in + WCNT, WCNT);
  k_f32_to_f16<<<(WCNT + 255) / 256, 256, 0, stream>>>(w_hh_f, w_hh16, WCNT);
  k_f32_to_f16<<<(WCNT + 255) / 256, 256, 0, stream>>>(w_hh_b, w_hh16 + WCNT, WCNT);

  dim3 ggrid(NROWS / 16, NCOLS / 256, 1);
  k_xp_gemm<<<ggrid, 128, 0, stream>>>(x16, w_in, b_ih_f, b_ih_b, xp);

  k_gru_scan<<<NBLK_SCAN, 32, 0, stream>>>(xp, w_hh16, b_hh_f, b_hh_b,
                                           input_lens, hf32, hf16, outb, ctr);

  k_segmax<<<B_ * NSEG_, 256, 0, stream>>>(outb, layout, input_lens,
                                           (float*)d_out);
  k_copy_h<<<(2 * B_ * H_ + 255) / 256, 256, 0, stream>>>(
      hf32, (float*)d_out + (size_t)B_ * NSEG_ * 2 * H_);
}
